// MLAttention_90013924590428
// MI455X (gfx1250) — matrix operand order repaired and checked by reading, not compiled, not run on hardware
//
#include <hip/hip_runtime.h>

// MLA forward for MI455X (gfx1250, wave32).
//  - f16 operands end-to-end (one-time f32->f16 casts), f32 WMMA accumulate
//  - weights pre-swizzled into WMMA B-fragment order -> fragment loads are
//    contiguous ds_load_b128 pairs
//  - GEMM A/B tiles staged by the Tensor Data Mover with double-buffered
//    ping-pong: next tile's TDM overlaps current tile's WMMAs; one barrier
//    per k-iteration. Flash V tiles also TDM-staged.
//  - matrix math exclusively on v_wmma_f32_16x16x32_f16

typedef __attribute__((ext_vector_type(16))) _Float16 v16h;
typedef __attribute__((ext_vector_type(8)))  float    v8f;
typedef unsigned int v4u __attribute__((ext_vector_type(4)));
typedef int          v8i __attribute__((ext_vector_type(8)));
typedef int          v4i __attribute__((ext_vector_type(4)));

static __device__ inline v8f v8f_zero() {
  v8f z = {0.f, 0.f, 0.f, 0.f, 0.f, 0.f, 0.f, 0.f};
  return z;
}

// K-slot permutation for 16x16x32 f16 B fragments (ISA 7.12.2).
// Slot o = hi*16 + j holds K = swz5(o); swz5 swaps bit3<->bit4 (involution).
static __device__ __host__ inline int swz5(int kk) {
  return ((kk & 8) << 1) | ((kk & 16) >> 1) | (kk & 7);
}

// Generic LDS pointer -> byte offset (ISA 10.2: LDS aperture uses addr[31:0])
static __device__ inline unsigned int lds_offset(const void* p) {
  return (unsigned int)(uintptr_t)p;
}

// ---------------------------------------------------------------------------
// Tensor Data Mover: 2D f16 tile load, global -> LDS, with row padding.
// D# packing per ISA 8.3 (group0) / 8.4 (group1); groups 2/3 zero (2D tile).
//   pad_interval code: interval = 2^(code+1) DWORDs (one full row here)
//   pad_amount  code: amount   = code+1 DWORDs appended per row
// ---------------------------------------------------------------------------
static __device__ inline void tdm_load_2d_f16(
    unsigned int lds_addr, const void* gptr,
    unsigned int tensor_d0, unsigned int tensor_d1,   // remaining extents (elems)
    unsigned int tile_d0, unsigned int tile_d1,       // tile size (elems)
    unsigned int d0_stride,                           // row stride (elems)
    unsigned int pad_interval_code, unsigned int pad_amount_code)
{
  const unsigned long long ga = (unsigned long long)(uintptr_t)gptr;
  v4u g0;
  g0[0] = 1u;                                           // count=1, user mode
  g0[1] = lds_addr;                                     // LDS byte address
  g0[2] = (unsigned int)ga;                             // global_addr[31:0]
  g0[3] = (unsigned int)((ga >> 32) & 0x01FFFFFFu) | (2u << 30);  // [56:32]|type=2

  const unsigned int w0 = (1u << 16)                    // data_size = 2 bytes
                        | (1u << 20)                    // pad_enable
                        | (pad_interval_code << 22)
                        | (pad_amount_code << 25);
  v8i g1;
  g1[0] = (int)w0;
  g1[1] = (int)((tensor_d0 & 0xFFFFu) << 16);           // dim0[15:0] (atomic=0)
  g1[2] = (int)(((tensor_d0 >> 16) & 0xFFFFu) | ((tensor_d1 & 0xFFFFu) << 16));
  g1[3] = (int)(((tensor_d1 >> 16) & 0xFFFFu) | ((tile_d0 & 0xFFFFu) << 16));
  g1[4] = (int)(tile_d1 & 0xFFFFu);                     // tile_dim2 = 0
  g1[5] = (int)d0_stride;                               // dim0_stride[31:0]
  g1[6] = 0;
  g1[7] = 0;
  v4i z4 = {0, 0, 0, 0};
#if defined(__clang_major__) && (__clang_major__ >= 23)
  v8i z8 = {0, 0, 0, 0, 0, 0, 0, 0};
  __builtin_amdgcn_tensor_load_to_lds(g0, g1, z4, z4, z8, 0);
#else
  __builtin_amdgcn_tensor_load_to_lds(g0, g1, z4, z4, 0);
#endif
}

// ---------------------------------------------------------------------------
// Elementwise f32 -> f16 cast (vectorized x4) — used for x only
// ---------------------------------------------------------------------------
__global__ __launch_bounds__(256) void cast_f32_to_f16(
    const float* __restrict__ s, _Float16* __restrict__ d, int n4)
{
  int i = blockIdx.x * 256 + threadIdx.x;
  if (i < n4) {
    const float4 v = ((const float4*)s)[i];
    d[4 * i + 0] = (_Float16)v.x;
    d[4 * i + 1] = (_Float16)v.y;
    d[4 * i + 2] = (_Float16)v.z;
    d[4 * i + 3] = (_Float16)v.w;
  }
}

// ---------------------------------------------------------------------------
// Weight cast + B-fragment swizzle: W (KxN f32, row-major) ->
// Wsw[kt][n][o] f16 with Wsw[kt][n][o] = W[kt*32 + swz5(o)][n].
// One block per (kt, o): reads one source row (coalesced), strided f16 writes.
// ---------------------------------------------------------------------------
__global__ __launch_bounds__(256) void cast_swizzle_w(
    const float* __restrict__ W, _Float16* __restrict__ Wsw, int N)
{
  const int kt = blockIdx.x >> 5;
  const int o  = blockIdx.x & 31;
  const int k  = kt * 32 + swz5(o);
  const float* src = W + (size_t)k * N;
  _Float16* dst = Wsw + (size_t)kt * N * 32 + o;
  for (int n = threadIdx.x; n < N; n += 256)
    dst[(size_t)n * 32] = (_Float16)src[n];
}

// ---------------------------------------------------------------------------
// Tiled WMMA GEMM: C(MxN) = A(MxK) @ Bsw; A f16 row-major, B pre-swizzled.
// Block tile 128x128, K-step 32, 256 threads = 8 waves (each 32x64).
// Double-buffered TDM staging: tile kt+1 in flight while computing tile kt.
// outMode: 0 = f32 row-major, 1 = f16 row-major, 2 = f16 V-swizzled
//          (element (m,n) -> [(m>>5)*ldc + n]*32 + swz5(m&31))
// ---------------------------------------------------------------------------
__global__ __launch_bounds__(256) void gemm_wmma_h(
    const _Float16* __restrict__ A, int lda,
    const _Float16* __restrict__ Bsw, int nb,   // nb = column count of W
    void* __restrict__ Cv, int ldc,
    int M, int N, int K, int outMode)
{
  __shared__ _Float16 As[2][128][40];   // 32 data halfs + TDM pad 4 DW
  __shared__ _Float16 Bs[2][128][40];   // row n: 32 swizzled K-halfs + pad

  const int tid  = threadIdx.x;
  const int lane = tid & 31;
  const int wid_s = __builtin_amdgcn_readfirstlane(tid >> 5);  // scalar wave id
  const int wm   = wid_s & 3;          // 4 waves along M (32 rows each)
  const int wn   = wid_s >> 2;         // 2 waves along N (64 cols each)
  const int l15  = lane & 15;
  const int hi   = lane >> 4;
  const int bm   = blockIdx.y * 128;
  const int bn   = blockIdx.x * 128;

  v8f acc[2][4];
#pragma unroll
  for (int i = 0; i < 2; ++i)
#pragma unroll
    for (int j = 0; j < 4; ++j) acc[i][j] = v8f_zero();

  const int ktiles = (K + 31) >> 5;

  // Prologue: stage tile 0 into buffer 0 (wave 0 only; scalar branch — TDM
  // ignores EXEC, so the guard must be a real branch).
  if (wid_s == 0) {
    tdm_load_2d_f16(lds_offset(&As[0][0][0]), &A[(size_t)bm * lda],
                    (unsigned)K, (unsigned)(M - bm), 32u, 128u,
                    (unsigned)lda, 3u, 3u);
    tdm_load_2d_f16(lds_offset(&Bs[0][0][0]), &Bsw[(size_t)bn * 32],
                    32u, (unsigned)(nb - bn), 32u, 128u, 32u, 3u, 3u);
    __builtin_amdgcn_s_wait_tensorcnt(0);
  }
  __syncthreads();

  for (int kt = 0; kt < ktiles; ++kt) {
    const int cur = kt & 1;
    // Kick off next tile into the other buffer; overlaps with WMMAs below.
    if (wid_s == 0 && kt + 1 < ktiles) {
      const int k1 = (kt + 1) << 5;
      tdm_load_2d_f16(lds_offset(&As[cur ^ 1][0][0]),
                      &A[(size_t)bm * lda + k1],
                      (unsigned)(K - k1), (unsigned)(M - bm),
                      32u, 128u, (unsigned)lda, 3u, 3u);
      tdm_load_2d_f16(lds_offset(&Bs[cur ^ 1][0][0]),
                      &Bsw[((size_t)(kt + 1) * nb + bn) * 32],
                      32u, (unsigned)(nb - bn), 32u, 128u, 32u, 3u, 3u);
    }

    // A fragment: lane row m = lane&15; 16 halfs = two aligned 16B chunks
    v16h af[2];
#pragma unroll
    for (int mt = 0; mt < 2; ++mt) {
      const int m = wm * 32 + mt * 16 + l15;
#pragma unroll
      for (int j = 0; j < 16; j += 2) {
        const int k = ((j & 8) << 1) + hi * 8 + (j & 6);
        af[mt][j]     = As[cur][m][k];
        af[mt][j + 1] = As[cur][m][k + 1];
      }
    }
    // B fragment: one contiguous 32B chunk per lane (pre-swizzled)
#pragma unroll
    for (int nt = 0; nt < 4; ++nt) {
      const v16h bf = *(const v16h*)&Bs[cur][wn * 64 + nt * 16 + l15][hi * 16];
#pragma unroll
      for (int mt = 0; mt < 2; ++mt)
        acc[mt][nt] = __builtin_amdgcn_wmma_f32_16x16x32_f16(
            false, af[mt], false, bf, (short)0, acc[mt][nt], false, false);
    }

    // Publish the prefetched tile: wave 0 drains its TENSORcnt, then the
    // block barrier orders it against everyone's reads next iteration.
    if (wid_s == 0) __builtin_amdgcn_s_wait_tensorcnt(0);
    __syncthreads();
  }

  // D layout: vgpr r -> row r (lanes 0-15) / r+8 (lanes 16-31); col = lane&15
#pragma unroll
  for (int mt = 0; mt < 2; ++mt)
#pragma unroll
    for (int nt = 0; nt < 4; ++nt)
#pragma unroll
      for (int r = 0; r < 8; ++r) {
        const int m = bm + wm * 32 + mt * 16 + r + hi * 8;
        const int n = bn + wn * 64 + nt * 16 + l15;
        if (m < M && n < N) {
          if (outMode == 0) {
            ((float*)Cv)[(size_t)m * ldc + n] = acc[mt][nt][r];
          } else if (outMode == 1) {
            ((_Float16*)Cv)[(size_t)m * ldc + n] = (_Float16)acc[mt][nt][r];
          } else {
            ((_Float16*)Cv)[((size_t)(m >> 5) * ldc + n) * 32 + swz5(m & 31)] =
                (_Float16)acc[mt][nt][r];
          }
        }
      }
}

// ---------------------------------------------------------------------------
// RMSNorm in place on f16 (f32 accumulation): one block per row
// ---------------------------------------------------------------------------
__global__ __launch_bounds__(256) void rmsnorm_f16(
    _Float16* __restrict__ X, const float* __restrict__ W, int ncols)
{
  __shared__ float red[256];
  const int row = blockIdx.x, tid = threadIdx.x;
  _Float16* p = X + (size_t)row * ncols;
  float ss = 0.f;
  for (int i = tid; i < ncols; i += 256) { const float v = (float)p[i]; ss += v * v; }
  red[tid] = ss;
  __syncthreads();
  for (int s = 128; s > 0; s >>= 1) {
    if (tid < s) red[tid] += red[tid + s];
    __syncthreads();
  }
  const float scale = rsqrtf(red[0] / (float)ncols + 1e-6f);
  for (int i = tid; i < ncols; i += 256)
    p[i] = (_Float16)((float)p[i] * scale * W[i]);
}

// ---------------------------------------------------------------------------
// Assemble q_cat/k_cat (T,H,192) f16: copy nope, RoPE rope parts (k broadcast)
// ---------------------------------------------------------------------------
__global__ __launch_bounds__(128) void build_qk_cat_h(
    const _Float16* __restrict__ q_nope,  // (T, 16*128)
    const _Float16* __restrict__ q_rope,  // (T, 16*64)
    const _Float16* __restrict__ k_nope,  // (T, 16*128)
    const _Float16* __restrict__ kvd,     // (T, 576), rope at [512:576]
    const float* __restrict__ cosp,       // (T, 32)
    const float* __restrict__ sinp,       // (T, 32)
    _Float16* __restrict__ q_cat,         // (T, 16, 192)
    _Float16* __restrict__ k_cat)         // (T, 16, 192)
{
  const int t = blockIdx.x >> 4;
  const int h = blockIdx.x & 15;
  const int tid = threadIdx.x;
  const size_t ob = ((size_t)t * 16 + h) * 192;
  q_cat[ob + tid] = q_nope[(size_t)t * 2048 + h * 128 + tid];
  k_cat[ob + tid] = k_nope[(size_t)t * 2048 + h * 128 + tid];
  if (tid < 32) {
    const float c = cosp[t * 32 + tid];
    const float s = sinp[t * 32 + tid];
    const _Float16* qr = q_rope + (size_t)t * 1024 + h * 64;
    const float xr = (float)qr[2 * tid], xi = (float)qr[2 * tid + 1];
    q_cat[ob + 128 + 2 * tid]     = (_Float16)(xr * c - xi * s);
    q_cat[ob + 128 + 2 * tid + 1] = (_Float16)(xr * s + xi * c);
    const _Float16* kr = kvd + (size_t)t * 576 + 512;
    const float kxr = (float)kr[2 * tid], kxi = (float)kr[2 * tid + 1];
    k_cat[ob + 128 + 2 * tid]     = (_Float16)(kxr * c - kxi * s);
    k_cat[ob + 128 + 2 * tid + 1] = (_Float16)(kxr * s + kxi * c);
  }
}

// ---------------------------------------------------------------------------
// Flash attention: grid (T/64, H), 128 threads = 4 waves, wave = 16 q-rows.
// Key tiles of 32. V tile staged by TDM from the swizzled v buffer; V and P
// fragments are contiguous per-lane chunks.
// ---------------------------------------------------------------------------
__global__ __launch_bounds__(128) void mla_flash_attn(
    const _Float16* __restrict__ Q,    // (T, H, 192)
    const _Float16* __restrict__ Kc,   // (T, H, 192)
    const _Float16* __restrict__ Vsw,  // swizzled: [T/32][2048][32]
    _Float16* __restrict__ O)          // (T, H*128) f16
{
  __shared__ _Float16 Qs[64][200];
  __shared__ _Float16 Ks[32][200];
  __shared__ _Float16 Vs[128][40];    // row = dv col, 32 swizzled key-halfs
  __shared__ _Float16 Ps[4][16][40];

  const int tid = threadIdx.x, lane = tid & 31;
  const int wid   = tid >> 5;
  const int wid_s = __builtin_amdgcn_readfirstlane(wid);  // scalar wave id
  const int l15 = lane & 15, hi = lane >> 4;
  const int h  = blockIdx.y;
  const int q0 = blockIdx.x * 64;

  // Stage Q tile (f16 copy, u32 chunks): 64 rows x 192 halfs
#pragma unroll 4
  for (int i = 0; i < 48; ++i) {
    const int e = i * 128 + tid;
    const int r = e / 96, c = e % 96;
    *(unsigned int*)&Qs[r][2 * c] =
        *(const unsigned int*)&Q[(((size_t)(q0 + r)) * 16 + h) * 192 + 2 * c];
  }

  v8f acc[8];
#pragma unroll
  for (int i = 0; i < 8; ++i) acc[i] = v8f_zero();
  float m_s[8], l_s[8];
#pragma unroll
  for (int r = 0; r < 8; ++r) { m_s[r] = -3.0e38f; l_s[r] = 0.f; }

  const float sc = 0.07216878364870322f;  // 192^-0.5
  const int qrow_base = q0 + wid * 16;
  const int nkt = 2 * blockIdx.x + 2;     // causal: keys <= q0+63

  for (int kb = 0; kb < nkt; ++kb) {
    const int k0 = kb * 32;
    // V tile via TDM: 128 rows (dv cols) x 32 swizzled key-halfs
    if (wid_s == 0) {
      tdm_load_2d_f16(lds_offset(&Vs[0][0]),
                      &Vsw[((size_t)kb * 2048 + h * 128) * 32],
                      32u, (unsigned)(2048 - h * 128),
                      32u, 128u, 32u, 3u, 3u);
    }
    // K tile (f16 copy, u32 chunks): 32 rows x 192 halfs
#pragma unroll 4
    for (int i = 0; i < 24; ++i) {
      const int e = i * 128 + tid;
      const int r = e / 96, c = e % 96;
      *(unsigned int*)&Ks[r][2 * c] =
          *(const unsigned int*)&Kc[(((size_t)(k0 + r)) * 16 + h) * 192 + 2 * c];
    }
    if (wid_s == 0) __builtin_amdgcn_s_wait_tensorcnt(0);
    __syncthreads();

    // S = Q K^T : 16x32 per wave, 6 WMMA k-steps over 192 dims
    v8f sa[2];
    sa[0] = v8f_zero();
    sa[1] = v8f_zero();
#pragma unroll
    for (int ks = 0; ks < 6; ++ks) {
      v16h aq;
      const int m = wid * 16 + l15;
#pragma unroll
      for (int j = 0; j < 16; j += 2) {
        const int k = ks * 32 + ((j & 8) << 1) + hi * 8 + (j & 6);
        aq[j]     = Qs[m][k];
        aq[j + 1] = Qs[m][k + 1];
      }
#pragma unroll
      for (int nt = 0; nt < 2; ++nt) {
        v16h bk;
        const int n = nt * 16 + l15;
#pragma unroll
        for (int j = 0; j < 16; ++j) bk[j] = Ks[n][ks * 32 + ((j & 8) << 1) + hi * 8 + (j & 7)];
        sa[nt] = __builtin_amdgcn_wmma_f32_16x16x32_f16(
            false, aq, false, bk, (short)0, sa[nt], false, false);
      }
    }

    // Scale + causal mask
#pragma unroll
    for (int nt = 0; nt < 2; ++nt)
#pragma unroll
      for (int r = 0; r < 8; ++r) {
        const float s = sa[nt][r] * sc;
        const int kidx = k0 + nt * 16 + l15;
        const int qrow = qrow_base + r + hi * 8;
        sa[nt][r] = (kidx > qrow) ? -1.0e9f : s;
      }

    // Online softmax (rows live across 16 lanes; xor 1/2/4/8 stays in half)
    float alpha[8];
#pragma unroll
    for (int r = 0; r < 8; ++r) {
      float v = fmaxf(sa[0][r], sa[1][r]);
      v = fmaxf(v, __shfl_xor(v, 1));
      v = fmaxf(v, __shfl_xor(v, 2));
      v = fmaxf(v, __shfl_xor(v, 4));
      v = fmaxf(v, __shfl_xor(v, 8));
      const float nm = fmaxf(m_s[r], v);
      alpha[r] = __expf(m_s[r] - nm);
      m_s[r] = nm;
    }

    float psum[8];
#pragma unroll
    for (int r = 0; r < 8; ++r) psum[r] = 0.f;
#pragma unroll
    for (int nt = 0; nt < 2; ++nt)
#pragma unroll
      for (int r = 0; r < 8; ++r) {
        const float p = __expf(sa[nt][r] - m_s[r]);
        psum[r] += p;
        // store P in A-fragment order: row m, slot position for key k
        const int kk = nt * 16 + l15;
        Ps[wid][r + hi * 8][swz5(kk)] = (_Float16)p;
      }
#pragma unroll
    for (int r = 0; r < 8; ++r) {
      float t = psum[r];
      t += __shfl_xor(t, 1);
      t += __shfl_xor(t, 2);
      t += __shfl_xor(t, 4);
      t += __shfl_xor(t, 8);
      l_s[r] = alpha[r] * l_s[r] + t;
    }
    __syncthreads();  // publish Ps; order against next-tile staging

    // O = alpha*O + P V : P fragment = one contiguous 32B chunk per lane
    const v16h ap = *(const v16h*)&Ps[wid][l15][hi * 16];
#pragma unroll
    for (int nt = 0; nt < 8; ++nt) {
#pragma unroll
      for (int r = 0; r < 8; ++r) acc[nt][r] *= alpha[r];
      const v16h bv = *(const v16h*)&Vs[nt * 16 + l15][hi * 16];
      acc[nt] = __builtin_amdgcn_wmma_f32_16x16x32_f16(
          false, ap, false, bv, (short)0, acc[nt], false, false);
    }
    __syncthreads();
  }

  // Epilogue: normalize, write f16 (T, H*128)
#pragma unroll
  for (int nt = 0; nt < 8; ++nt)
#pragma unroll
    for (int r = 0; r < 8; ++r) {
      const int row = qrow_base + r + hi * 8;
      O[(size_t)row * 2048 + h * 128 + nt * 16 + l15] =
          (_Float16)(acc[nt][r] / l_s[r]);
    }
}

// ---------------------------------------------------------------------------
static inline void launch_gemm(const _Float16* A, int lda, const _Float16* Bsw,
                               int nb, void* C, int ldc, int M, int N, int K,
                               int outMode, hipStream_t s)
{
  dim3 g((N + 127) / 128, (M + 127) / 128);
  gemm_wmma_h<<<g, 256, 0, s>>>(A, lda, Bsw, nb, C, ldc, M, N, K, outMode);
}

static inline void launch_cast(const float* s, _Float16* d, size_t n, hipStream_t st)
{
  const int n4 = (int)(n / 4);
  cast_f32_to_f16<<<(n4 + 255) / 256, 256, 0, st>>>(s, d, n4);
}

static inline void launch_castw(const float* w, _Float16* wsw, int K, int N,
                                hipStream_t st)
{
  cast_swizzle_w<<<K, 256, 0, st>>>(w, wsw, N);   // K blocks = (K/32)*32
}

extern "C" void kernel_launch(void* const* d_in, const int* in_sizes, int n_in,
                              void* d_out, int out_size, void* d_ws, size_t ws_size,
                              hipStream_t stream)
{
  (void)in_sizes; (void)n_in; (void)out_size;
  const float* x           = (const float*)d_in[0];
  const float* fcos        = (const float*)d_in[1];
  const float* fsin        = (const float*)d_in[2];
  // d_in[3] = mask: causal mask applied analytically in the flash kernel
  const float* Wq_down     = (const float*)d_in[4];
  const float* q_norm_w    = (const float*)d_in[5];
  const float* Wq_up_nope  = (const float*)d_in[6];
  const float* Wq_up_rope  = (const float*)d_in[7];
  const float* Wkv_down    = (const float*)d_in[8];
  const float* Wkv_up_nope = (const float*)d_in[9];
  const float* Wkv_up_v    = (const float*)d_in[10];
  const float* Wo          = (const float*)d_in[11];
  float* out = (float*)d_out;
  _Float16* ws = (_Float16*)d_ws;

  size_t o = 0;
  _Float16* x_h    = ws + o; o += (size_t)2048 * 2048;
  _Float16* wqd    = ws + o; o += (size_t)2048 * 1536;   // swizzled
  _Float16* wqun   = ws + o; o += (size_t)1536 * 2048;   // swizzled
  _Float16* wqur   = ws + o; o += (size_t)1536 * 1024;   // swizzled
  _Float16* wkvd   = ws + o; o += (size_t)2048 * 576;    // swizzled
  _Float16* wkun   = ws + o; o += (size_t)512 * 2048;    // swizzled
  _Float16* wkuv   = ws + o; o += (size_t)512 * 2048;    // swizzled
  _Float16* wo_h   = ws + o; o += (size_t)2048 * 2048;   // swizzled
  _Float16* qd_h   = ws + o; o += (size_t)2048 * 1536;
  _Float16* q_nope = ws + o; o += (size_t)2048 * 2048;
  _Float16* q_rope = ws + o; o += (size_t)2048 * 1024;
  _Float16* kv_h   = ws + o; o += (size_t)2048 * 576;
  _Float16* k_nope = ws + o; o += (size_t)2048 * 2048;
  _Float16* v_sw   = ws + o; o += (size_t)2048 * 2048;   // V-swizzled
  _Float16* q_cat  = ws + o; o += (size_t)2048 * 16 * 192;
  _Float16* k_cat  = ws + o; o += (size_t)2048 * 16 * 192;
  _Float16* attn_h = q_nope;  // q_nope dead after build_qk_cat; reuse
  if (ws_size < o * sizeof(_Float16)) return;

  // One-time casts: x row-major; weights cast + B-fragment swizzle
  launch_cast(x, x_h, (size_t)2048 * 2048, stream);
  launch_castw(Wq_down,     wqd,  2048, 1536, stream);
  launch_castw(Wq_up_nope,  wqun, 1536, 2048, stream);
  launch_castw(Wq_up_rope,  wqur, 1536, 1024, stream);
  launch_castw(Wkv_down,    wkvd, 2048, 576,  stream);
  launch_castw(Wkv_up_nope, wkun, 512,  2048, stream);
  launch_castw(Wkv_up_v,    wkuv, 512,  2048, stream);
  launch_castw(Wo,          wo_h, 2048, 2048, stream);

  // qd = rmsnorm(x @ Wq_down)
  launch_gemm(x_h, 2048, wqd, 1536, qd_h, 1536, 2048, 1536, 2048, 1, stream);
  rmsnorm_f16<<<2048, 256, 0, stream>>>(qd_h, q_norm_w, 1536);
  // q projections
  launch_gemm(qd_h, 1536, wqun, 2048, q_nope, 2048, 2048, 2048, 1536, 1, stream);
  launch_gemm(qd_h, 1536, wqur, 1024, q_rope, 1024, 2048, 1024, 1536, 1, stream);
  // kv down + up (c_kv = kv[:, :512] -> lda=576, K=512)
  launch_gemm(x_h, 2048, wkvd, 576, kv_h, 576, 2048, 576, 2048, 1, stream);
  launch_gemm(kv_h, 576, wkun, 2048, k_nope, 2048, 2048, 2048, 512, 1, stream);
  launch_gemm(kv_h, 576, wkuv, 2048, v_sw,   2048, 2048, 2048, 512, 2, stream);
  // rope + concat
  build_qk_cat_h<<<2048 * 16, 128, 0, stream>>>(q_nope, q_rope, k_nope, kv_h,
                                                fcos, fsin, q_cat, k_cat);
  // flash attention (V from swizzled buffer; writes f16)
  mla_flash_attn<<<dim3(2048 / 64, 16), 128, 0, stream>>>(q_cat, k_cat, v_sw, attn_h);
  // output projection -> f32 d_out
  launch_gemm(attn_h, 2048, wo_h, 2048, out, 2048, 2048, 2048, 2048, 0, stream);
}
